// DihedralMessagePassing_34093450396331
// MI455X (gfx1250) — matrix operand order repaired and checked by reading, not compiled, not run on hardware
//
#include <hip/hip_runtime.h>
#include <hip/hip_bf16.h>
#include <stdint.h>

typedef __bf16 bf16_t;
typedef __attribute__((ext_vector_type(16))) __bf16 v16bf;
typedef __attribute__((ext_vector_type(8)))  float  v8f;

// CDNA5 split wait counters.
#define WAVE_SYNC()   asm volatile("s_wait_dscnt 0" ::: "memory")
#define WAIT_ASYNC(n) asm volatile("s_wait_asynccnt " #n ::: "memory")

// A-operand k order used by this code (16x32): lane L holds row m = L&15.
// Element e: K = 16*(L>>4) + e (contiguous, matches B fill).
__device__ __forceinline__ int a_kmap(int lane, int e) {
  int half = lane >> 4;
  return (half << 4) + e;
}

__device__ __forceinline__ v8f zero8() {
  v8f z;
#pragma unroll
  for (int i = 0; i < 8; ++i) z[i] = 0.0f;
  return z;
}

__device__ __forceinline__ float silu(float h) {
  return h / (1.0f + __expf(-h));
}

// Async DMA of one 256B row segment: global bf16 -> LDS, no VGPR round-trip.
__device__ __forceinline__ void async_row_b64(unsigned lds_off, uint64_t gaddr) {
  asm volatile("global_load_async_to_lds_b64 %0, %1, off"
               :: "v"(lds_off), "v"(gaddr) : "memory");
}

// Stage one 16-dihedral tile's [x_i | x_l] block (16 x 256 bf16) via 32 async ops.
__device__ __forceinline__ void stage_tile_async(unsigned abase, uint64_t xbase,
                                                 int iv, int lv, int lane) {
#pragma unroll
  for (int r = 0; r < 16; ++r) {
    const int ivr = __shfl(iv, r);
    const int lvr = __shfl(lv, r);
    const unsigned la = abase + (unsigned)(r * 576 + lane * 8);
    async_row_b64(la,       xbase + ((uint64_t)(uint32_t)ivr << 8) + (unsigned)(lane * 8));
    async_row_b64(la + 256, xbase + ((uint64_t)(uint32_t)lvr << 8) + (unsigned)(lane * 8));
  }
}

// ---------------------------------------------------------------- utilities
__global__ void zero_f32_kernel(float* __restrict__ p, long n) {
  long i = (long)blockIdx.x * blockDim.x + threadIdx.x;
  long stride = (long)gridDim.x * blockDim.x;
  for (; i < n; i += stride) p[i] = 0.0f;
}

__global__ void cvt_x_kernel(const float* __restrict__ x, bf16_t* __restrict__ x_bf, long n) {
  long i = (long)blockIdx.x * blockDim.x + threadIdx.x;
  long stride = (long)gridDim.x * blockDim.x;
  for (; i < n; i += stride) x_bf[i] = (bf16_t)x[i];
}

// Transpose weights to column-major bf16 (W1 K-padded 259->288 with zeros).
__global__ void prep_weights_kernel(const float* __restrict__ W1,
                                    const float* __restrict__ W2,
                                    const float* __restrict__ W3,
                                    const float* __restrict__ Wout,
                                    bf16_t* __restrict__ W1t,   // 64 cols x 288 rows
                                    bf16_t* __restrict__ W2t,   // 64 cols x 64 rows
                                    bf16_t* __restrict__ W3t,   // 128 cols x 64 rows
                                    bf16_t* __restrict__ WoutT) // 128 cols x 128 rows
{
  const int S1 = 64 * 288, S2 = 64 * 64, S3 = 128 * 64, S4 = 128 * 128;
  int gid = blockIdx.x * blockDim.x + threadIdx.x;
  if (gid < S1) {
    int col = gid / 288, row = gid % 288;
    W1t[gid] = (row < 259) ? (bf16_t)W1[row * 64 + col] : (bf16_t)0.0f;
  } else if (gid < S1 + S2) {
    int g = gid - S1; int col = g / 64, row = g % 64;
    W2t[g] = (bf16_t)W2[row * 64 + col];
  } else if (gid < S1 + S2 + S3) {
    int g = gid - S1 - S2; int col = g / 64, row = g % 64;
    W3t[g] = (bf16_t)W3[row * 128 + col];
  } else if (gid < S1 + S2 + S3 + S4) {
    int g = gid - S1 - S2 - S3; int col = g / 128, row = g % 128;
    WoutT[g] = (bf16_t)Wout[row * 128 + col];
  }
}

__global__ void count_kernel(const int* __restrict__ qidx,
                             float* __restrict__ cnt_j, float* __restrict__ cnt_k,
                             int n_dih) {
  int d = blockIdx.x * blockDim.x + threadIdx.x;
  if (d < n_dih) {
    atomicAdd(&cnt_j[qidx[(size_t)n_dih + d]], 1.0f);
    atomicAdd(&cnt_k[qidx[(size_t)2 * n_dih + d]], 1.0f);
  }
}

// Per-tile pipeline stage: prefetch next tile (async DMA), run the 3-GEMM MLP
// on the current statically-addressed LDS buffer, scatter results.
__device__ __forceinline__ void process_tile(
    bf16_t* __restrict__ A,          // current tile's staged input (static LDS ptr)
    unsigned next_abase,             // raw LDS offset of the OTHER buffer
    bf16_t* __restrict__ Hbuf, bf16_t* __restrict__ H2buf,
    const bf16_t* __restrict__ W1t, const bf16_t* __restrict__ W2t,
    const bf16_t* __restrict__ W3t,
    const int* __restrict__ qidx, const float* __restrict__ attr,
    const float* __restrict__ b1, const float* __restrict__ b2,
    const float* __restrict__ b3,
    float* __restrict__ sum_j, float* __restrict__ sum_k,
    uint64_t xbase, int n_dih, int nTiles, int stride,
    int lane, int half, int nloc,
    int& tile, bool& have, int& iv, int& lv, int& jv, int& kv)
{
  const int d0 = tile << 4;
  const int ntile = tile + stride;
  const bool have_next = (ntile < nTiles);

  // Issue next tile's async stage, then wait until the CURRENT tile's 32
  // async ops (issued earlier, in-order counter) have completed.
  int niv = 0, nlv = 0, njv = -1, nkv = -1;
  if (have_next) {
    const int nd = (ntile << 4) + lane;
    if (lane < 16 && nd < n_dih) {
      niv = qidx[nd];
      njv = qidx[(size_t)n_dih + nd];
      nkv = qidx[(size_t)2 * n_dih + nd];
      nlv = qidx[(size_t)3 * n_dih + nd];
    }
    stage_tile_async(next_abase, xbase, niv, nlv, lane);
    WAIT_ASYNC(0x20);
  } else {
    WAIT_ASYNC(0);
  }

  // Stage attr (cols 256..258) for the current tile.
  if (lane < 16) {
    const int d = d0 + lane;
    const bool dv = (d < n_dih);
#pragma unroll
    for (int q = 0; q < 3; ++q)
      A[lane * 288 + 256 + q] = dv ? (bf16_t)attr[(size_t)d * 3 + q] : (bf16_t)0.0f;
  }
  WAVE_SYNC();

  // ---- GEMM1: (16x288)x(288x64), 9 K-chunks, 4 N-tiles
  v8f acc[4];
#pragma unroll
  for (int nt = 0; nt < 4; ++nt) acc[nt] = zero8();
  {
    const bf16_t* arow = A + nloc * 288;
#pragma unroll
    for (int c = 0; c < 9; ++c) {
      v16bf af;
#pragma unroll
      for (int e = 0; e < 16; ++e) af[e] = arow[c * 32 + a_kmap(lane, e)];
#pragma unroll
      for (int nt = 0; nt < 4; ++nt) {
        const bf16_t* bp = W1t + (nt * 16 + nloc) * 288 + c * 32 + (half << 4);
        v16bf bf;
#pragma unroll
        for (int e = 0; e < 16; ++e) bf[e] = bp[e];
        acc[nt] = __builtin_amdgcn_wmma_f32_16x16x32_bf16(
            false, af, false, bf, (short)0, acc[nt], false, false);
      }
    }
  }
  // SiLU -> Hbuf (A layout source for GEMM2)
#pragma unroll
  for (int nt = 0; nt < 4; ++nt) {
    const int n = nt * 16 + nloc;
    const float bias = b1[n];
#pragma unroll
    for (int v = 0; v < 8; ++v) {
      const int m = v + (half << 3);
      Hbuf[m * 64 + n] = (bf16_t)silu(acc[nt][v] + bias);
    }
  }
  WAVE_SYNC();

  // ---- GEMM2: (16x64)x(64x64), 2 K-chunks, 4 N-tiles
#pragma unroll
  for (int nt = 0; nt < 4; ++nt) acc[nt] = zero8();
  {
    const bf16_t* hrow = Hbuf + nloc * 64;
#pragma unroll
    for (int c = 0; c < 2; ++c) {
      v16bf af;
#pragma unroll
      for (int e = 0; e < 16; ++e) af[e] = hrow[c * 32 + a_kmap(lane, e)];
#pragma unroll
      for (int nt = 0; nt < 4; ++nt) {
        const bf16_t* bp = W2t + (nt * 16 + nloc) * 64 + c * 32 + (half << 4);
        v16bf bf;
#pragma unroll
        for (int e = 0; e < 16; ++e) bf[e] = bp[e];
        acc[nt] = __builtin_amdgcn_wmma_f32_16x16x32_bf16(
            false, af, false, bf, (short)0, acc[nt], false, false);
      }
    }
  }
#pragma unroll
  for (int nt = 0; nt < 4; ++nt) {
    const int n = nt * 16 + nloc;
    const float bias = b2[n];
#pragma unroll
    for (int v = 0; v < 8; ++v) {
      const int m = v + (half << 3);
      H2buf[m * 64 + n] = (bf16_t)silu(acc[nt][v] + bias);
    }
  }
  WAVE_SYNC();

  // ---- GEMM3: (16x64)x(64x128), 2 K-chunks, 8 N-tiles
  v8f acc3[8];
#pragma unroll
  for (int nt = 0; nt < 8; ++nt) acc3[nt] = zero8();
  {
    const bf16_t* hrow = H2buf + nloc * 64;
#pragma unroll
    for (int c = 0; c < 2; ++c) {
      v16bf af;
#pragma unroll
      for (int e = 0; e < 16; ++e) af[e] = hrow[c * 32 + a_kmap(lane, e)];
#pragma unroll
      for (int nt = 0; nt < 8; ++nt) {
        const bf16_t* bp = W3t + (nt * 16 + nloc) * 64 + c * 32 + (half << 4);
        v16bf bf;
#pragma unroll
        for (int e = 0; e < 16; ++e) bf[e] = bp[e];
        acc3[nt] = __builtin_amdgcn_wmma_f32_16x16x32_bf16(
            false, af, false, bf, (short)0, acc3[nt], false, false);
      }
    }
  }

  // Scatter-add msgs into L2-resident accumulators (fp32 global atomics).
#pragma unroll
  for (int v = 0; v < 8; ++v) {
    const int m = v + (half << 3);
    const int jn = __shfl(jv, m);
    const int kn = __shfl(kv, m);
    if (jn >= 0) {
#pragma unroll
      for (int nt = 0; nt < 8; ++nt) {
        const int n = nt * 16 + nloc;
        const float val = acc3[nt][v] + b3[n];
        atomicAdd(&sum_j[(size_t)jn * 128 + n], val);
        atomicAdd(&sum_k[(size_t)kn * 128 + n], val);
      }
    }
  }

  iv = niv; lv = nlv; jv = njv; kv = nkv;
  tile = ntile; have = have_next;
}

// ------------------------------------------------- phase 1: MLP + scatter-add
__global__ void __launch_bounds__(256)
mlp_scatter_kernel(const bf16_t* __restrict__ x_bf, const int* __restrict__ qidx,
                   const float* __restrict__ attr,
                   const bf16_t* __restrict__ Wt_g,  // W1t|W2t|W3t contiguous
                   const float* __restrict__ b1, const float* __restrict__ b2,
                   const float* __restrict__ b3,
                   float* __restrict__ sum_j, float* __restrict__ sum_k,
                   int n_dih)
{
  extern __shared__ char smem[];
  bf16_t* W1t = (bf16_t*)smem;      // 64 x 288
  bf16_t* W2t = W1t + 64 * 288;     // 64 x 64
  bf16_t* W3t = W2t + 64 * 64;      // 128 x 64
  char* pw = (char*)(W3t + 128 * 64);

  const int wave = threadIdx.x >> 5;
  const int lane = threadIdx.x & 31;
  const int half = lane >> 4;
  const int nloc = lane & 15;

  const int PW = (16 * 288 * 2 + 16 * 64 * 2) * 2;  // 22528 B per wave
  char* mine = pw + wave * PW;
  bf16_t* Abuf0 = (bf16_t*)mine;         // static LDS pointers: keep DS path
  bf16_t* Abuf1 = Abuf0 + 16 * 288;
  bf16_t* Hbuf  = Abuf1 + 16 * 288;      // 16 x 64 hidden 1
  bf16_t* H2buf = Hbuf + 16 * 64;        // 16 x 64 hidden 2

  // Cooperative weight stage: 30720 bf16 elements.
  for (int t = threadIdx.x; t < 64 * 288 + 64 * 64 + 128 * 64; t += blockDim.x)
    W1t[t] = Wt_g[t];

  // Zero pad columns 256..287 of both A buffers once (attr overwrites 256..258).
  for (int t = lane; t < 512; t += 32) {
    const int r = t >> 5, c = 256 + (t & 31);
    Abuf0[r * 288 + c] = (bf16_t)0.0f;
    Abuf1[r * 288 + c] = (bf16_t)0.0f;
  }
  __syncthreads();

  const uint64_t xbase  = (uint64_t)(uintptr_t)x_bf;
  const unsigned abase0 = (unsigned)(uintptr_t)Abuf0;  // raw LDS offsets for async dst
  const unsigned abase1 = (unsigned)(uintptr_t)Abuf1;

  const int nTiles = (n_dih + 15) >> 4;
  const int stride = gridDim.x * 8;
  int tile = blockIdx.x * 8 + wave;

  // Prologue: load indices + launch async stage for the first tile into buf0.
  int iv = 0, lv = 0, jv = -1, kv = -1;
  bool have = (tile < nTiles);
  if (have) {
    const int d = (tile << 4) + lane;
    if (lane < 16 && d < n_dih) {
      iv = qidx[d];
      jv = qidx[(size_t)n_dih + d];
      kv = qidx[(size_t)2 * n_dih + d];
      lv = qidx[(size_t)3 * n_dih + d];
    }
    stage_tile_async(abase0, xbase, iv, lv, lane);
  }

  // Two tiles per iteration: static buffer pointers, no dynamic select.
  while (have) {
    process_tile(Abuf0, abase1, Hbuf, H2buf, W1t, W2t, W3t, qidx, attr,
                 b1, b2, b3, sum_j, sum_k, xbase, n_dih, nTiles, stride,
                 lane, half, nloc, tile, have, iv, lv, jv, kv);
    if (!have) break;
    process_tile(Abuf1, abase0, Hbuf, H2buf, W1t, W2t, W3t, qidx, attr,
                 b1, b2, b3, sum_j, sum_k, xbase, n_dih, nTiles, stride,
                 lane, half, nloc, tile, have, iv, lv, jv, kv);
  }
}

// ------------------------------------------------- phase 2: mean + out GEMM
__global__ void __launch_bounds__(256)
node_out_kernel(const float* __restrict__ sum_j, const float* __restrict__ sum_k,
                const float* __restrict__ cnt_j, const float* __restrict__ cnt_k,
                const bf16_t* __restrict__ WoutT_g, float* __restrict__ out,
                int n_nodes)
{
  extern __shared__ char smem[];
  bf16_t* Wt = (bf16_t*)smem;  // 128 cols x 128 rows (column-major)
  const int wave = threadIdx.x >> 5;
  const int lane = threadIdx.x & 31;
  const int half = lane >> 4;
  const int nloc = lane & 15;
  bf16_t* Agg = (bf16_t*)(Wt + 128 * 128) + wave * (16 * 128);

  for (int t = threadIdx.x; t < 128 * 128; t += blockDim.x) Wt[t] = WoutT_g[t];
  __syncthreads();

  const int nTiles = (n_nodes + 15) >> 4;
  for (int tile = blockIdx.x * 8 + wave; tile < nTiles; tile += gridDim.x * 8) {
    const int node0 = tile << 4;
    WAVE_SYNC();
    for (int r = 0; r < 16; ++r) {
      const int nd = node0 + r;
      if (nd < n_nodes) {
        const float rj = 1.0f / fmaxf(cnt_j[nd], 1.0f);
        const float rk = 1.0f / fmaxf(cnt_k[nd], 1.0f);
#pragma unroll
        for (int tc = 0; tc < 4; ++tc) {
          const int c = tc * 32 + lane;
          const float v = (sum_j[(size_t)nd * 128 + c] * rj +
                           sum_k[(size_t)nd * 128 + c] * rk) * 0.5f;
          Agg[r * 128 + c] = (bf16_t)v;
        }
      } else {
#pragma unroll
        for (int tc = 0; tc < 4; ++tc) Agg[r * 128 + tc * 32 + lane] = (bf16_t)0.0f;
      }
    }
    WAVE_SYNC();

    v8f acc[8];
#pragma unroll
    for (int nt = 0; nt < 8; ++nt) acc[nt] = zero8();
    const bf16_t* arow = Agg + nloc * 128;
#pragma unroll
    for (int c = 0; c < 4; ++c) {
      v16bf af;
#pragma unroll
      for (int e = 0; e < 16; ++e) af[e] = arow[c * 32 + a_kmap(lane, e)];
#pragma unroll
      for (int nt = 0; nt < 8; ++nt) {
        const bf16_t* bp = Wt + (nt * 16 + nloc) * 128 + c * 32 + (half << 4);
        v16bf bf;
#pragma unroll
        for (int e = 0; e < 16; ++e) bf[e] = bp[e];
        acc[nt] = __builtin_amdgcn_wmma_f32_16x16x32_bf16(
            false, af, false, bf, (short)0, acc[nt], false, false);
      }
    }
#pragma unroll
    for (int v = 0; v < 8; ++v) {
      const int nd = node0 + v + (half << 3);
      if (nd < n_nodes) {
#pragma unroll
        for (int nt = 0; nt < 8; ++nt)
          out[(size_t)nd * 128 + nt * 16 + nloc] = acc[nt][v];
      }
    }
  }
}

// --------------------------------------------------------------- launcher
extern "C" void kernel_launch(void* const* d_in, const int* in_sizes, int n_in,
                              void* d_out, int out_size, void* d_ws, size_t ws_size,
                              hipStream_t stream)
{
  const float* x    = (const float*)d_in[0];
  const int*   qidx = (const int*)d_in[1];
  const float* attr = (const float*)d_in[2];
  const float* W1   = (const float*)d_in[3];
  const float* b1   = (const float*)d_in[4];
  const float* W2   = (const float*)d_in[5];
  const float* b2   = (const float*)d_in[6];
  const float* W3   = (const float*)d_in[7];
  const float* b3   = (const float*)d_in[8];
  const float* Wout = (const float*)d_in[9];
  float* out = (float*)d_out;

  const int n_nodes = in_sizes[0] / 128;
  const int n_dih   = in_sizes[1] / 4;

  char* ws = (char*)d_ws;
  size_t off = 0;
  float* sum_j = (float*)(ws + off); off += (size_t)n_nodes * 128 * 4;
  float* sum_k = (float*)(ws + off); off += (size_t)n_nodes * 128 * 4;
  float* cnt_j = (float*)(ws + off); off += (size_t)n_nodes * 4;
  float* cnt_k = (float*)(ws + off); off += (size_t)n_nodes * 4;
  off = (off + 255) & ~(size_t)255;
  bf16_t* W1t   = (bf16_t*)(ws + off); off += (size_t)64 * 288 * 2;
  bf16_t* W2t   = (bf16_t*)(ws + off); off += (size_t)64 * 64 * 2;
  bf16_t* W3t   = (bf16_t*)(ws + off); off += (size_t)128 * 64 * 2;
  bf16_t* WoutT = (bf16_t*)(ws + off); off += (size_t)128 * 128 * 2;
  off = (off + 255) & ~(size_t)255;
  bf16_t* x_bf  = (bf16_t*)(ws + off); off += (size_t)n_nodes * 128 * 2;

  // 1) zero accumulators (sum_j|sum_k|cnt_j|cnt_k are contiguous)
  const long nz = (long)n_nodes * 128 * 2 + (long)n_nodes * 2;
  zero_f32_kernel<<<2048, 256, 0, stream>>>(sum_j, nz);

  // 2) weight transpose/convert + x conversion to bf16
  const int prepN = 64 * 288 + 64 * 64 + 128 * 64 + 128 * 128;
  prep_weights_kernel<<<(prepN + 255) / 256, 256, 0, stream>>>(
      W1, W2, W3, Wout, W1t, W2t, W3t, WoutT);
  cvt_x_kernel<<<2048, 256, 0, stream>>>(x, x_bf, (long)n_nodes * 128);

  // 3) per-node counts
  count_kernel<<<(n_dih + 255) / 256, 256, 0, stream>>>(qidx, cnt_j, cnt_k, n_dih);

  // 4) per-dihedral MLP + scatter (async double-buffered staging)
  const size_t smem1 = (size_t)(64 * 288 + 64 * 64 + 128 * 64) * 2 +
                       8 * (size_t)((16 * 288 * 2 + 16 * 64 * 2) * 2);
  (void)hipFuncSetAttribute((const void*)mlp_scatter_kernel,
                            hipFuncAttributeMaxDynamicSharedMemorySize, (int)smem1);
  mlp_scatter_kernel<<<768, 256, smem1, stream>>>(
      x_bf, qidx, attr, W1t, b1, b2, b3, sum_j, sum_k, n_dih);

  // 5) scatter-mean finalize + output GEMM
  const size_t smem2 = (size_t)128 * 128 * 2 + 8 * (size_t)16 * 128 * 2;
  (void)hipFuncSetAttribute((const void*)node_out_kernel,
                            hipFuncAttributeMaxDynamicSharedMemorySize, (int)smem2);
  node_out_kernel<<<160, 256, smem2, stream>>>(
      sum_j, sum_k, cnt_j, cnt_k, WoutT, out, n_nodes);
}
